// ResidualMemoryModel_45561013076297
// MI455X (gfx1250) — matrix operand order repaired and checked by reading, not compiled, not run on hardware
//
#include <hip/hip_runtime.h>

typedef __bf16 bf16;
typedef __attribute__((ext_vector_type(16))) __bf16 v16bf;
typedef __attribute__((ext_vector_type(8)))  __bf16 v8bf;
typedef __attribute__((ext_vector_type(8)))  float   v8f;

#define TT 128
#define BB 16
#define HH 512
#define VV 50257
#define LL 6
#define VPAD 50304        // 393 * 128
#define LROW 520          // padded LDS row (bf16 elems); 520*2=1040B, 16B aligned, banks spread

__device__ __forceinline__ v16bf ldfrag(const bf16* p0, const bf16* p1) {
  v8bf lo = *(const v8bf*)p0;
  v8bf hi = *(const v8bf*)p1;
  return __builtin_shufflevector(lo, hi, 0,1,2,3,4,5,6,7,8,9,10,11,12,13,14,15);
}

// ---------------- prep kernels ----------------

__global__ void __launch_bounds__(256) prep_w_kernel(
    const float* __restrict__ Cw, const float* __restrict__ Pw,
    bf16* __restrict__ cwb, bf16* __restrict__ pwb) {
  int idx = blockIdx.x * 256 + threadIdx.x;
  const int n = LL * HH * HH;
  if (idx < n)            cwb[idx]     = (bf16)Cw[idx];
  else if (idx < 2 * n)   pwb[idx - n] = (bf16)Pw[idx - n];
}

__global__ void __launch_bounds__(256) prep_dw_kernel(
    const float* __restrict__ Dw, bf16* __restrict__ dwb) {
  size_t idx = (size_t)blockIdx.x * 256 + threadIdx.x;
  if (idx >= (size_t)VPAD * HH) return;
  size_t v = idx >> 9;
  dwb[idx] = (v < VV) ? (bf16)Dw[idx] : (bf16)0.0f;
}

// ring layout: ring[i][slot][b][h], slot = timestep & 7 (x_i(t) at slot t&7; t<0 prefilled)
__global__ void __launch_bounds__(256) prep_ring_kernel(
    const float* __restrict__ h0, const float* __restrict__ h1,
    const float* __restrict__ h2, const float* __restrict__ h3,
    const float* __restrict__ h4, const float* __restrict__ h5,
    bf16* __restrict__ ring) {
  int idx = blockIdx.x * 256 + threadIdx.x;
  if (idx >= 21 * BB * HH) return;
  int h = idx & (HH - 1);
  int rem = idx >> 9;
  int b = rem & (BB - 1);
  int slot = rem >> 4;            // 0..20 enumerates (i,j)
  int i = 0, j = slot;
  while (j >= i + 1) { j -= (i + 1); i++; }
  const float* hs[6] = {h0, h1, h2, h3, h4, h5};
  float v = hs[i][((size_t)j * BB + b) * HH + h];
  int rs = (7 - j) & 7;           // tau = -1-j  ->  slot (tau & 7)
  ring[(((size_t)i * 8 + rs) * BB + b) * HH + h] = (bf16)v;
}

// ---------------- recurrent scan: 1 workgroup, 32 waves ----------------

__global__ void __launch_bounds__(1024) recurrent_kernel(
    const int*   __restrict__ tokens, const float* __restrict__ embW,
    const float* __restrict__ Cb,     const float* __restrict__ Pb,
    const bf16*  __restrict__ cw,     const bf16*  __restrict__ pw,
    bf16* __restrict__ ring, bf16* __restrict__ tops) {
  __shared__ bf16 xcur[16 * LROW];   // current layer input x_i(t), [16][512] bf16
  __shared__ bf16 xh  [16 * LROW];   // hist_h_projected staging
  const int tid  = threadIdx.x;
  const int lane = tid & 31;
  const int wv   = tid >> 5;         // 0..31 : N-tile index
  const int nl   = lane & 15;
  const int hh   = lane >> 4;
  const int ncol = wv * 16 + nl;

  for (int t = 0; t < TT; ++t) {
    // x0 = emb_W[tokens[t]]
    for (int idx = tid; idx < BB * HH; idx += 1024) {
      int b = idx >> 9, h = idx & (HH - 1);
      int tok = tokens[t * BB + b];
      bf16 bv = (bf16)embW[(size_t)tok * HH + h];
      xcur[b * LROW + h] = bv;
      ring[((size_t)(t & 7) * BB + b) * HH + h] = bv;   // layer 0 ring
    }
    __syncthreads();

    for (int i = 0; i < LL; ++i) {
      const int tau = t - 1 - i;    // bitwise & handles negatives mod 8
      const bf16* histA = ring + ((size_t)(i * 8 + (tau & 7)) * BB) * HH;
      const bf16* cwi = cw + (size_t)i * HH * HH + (size_t)ncol * HH;
      const bf16* pwi = pw + (size_t)i * HH * HH + (size_t)ncol * HH;

      float cb = Cb[i * HH + ncol];
      v8f acc0, acc1;
#pragma unroll
      for (int r = 0; r < 8; ++r) { acc0[r] = cb; acc1[r] = cb; }

      // pass 1: stacked A = [x_i(t) ; x_i(t-1-i)] times Cw[i].T  (Cw tile loaded once)
#pragma unroll 4
      for (int k = 0; k < HH; k += 32) {
        v16bf a0 = ldfrag(&xcur[nl * LROW + k + hh * 8], &xcur[nl * LROW + k + 16 + hh * 8]);
        v16bf a1 = ldfrag(&histA[(size_t)nl * HH + k + hh * 8],
                          &histA[(size_t)nl * HH + k + 16 + hh * 8]);
        v16bf b0 = ldfrag(&cwi[k + hh * 8], &cwi[k + 16 + hh * 8]);
        acc0 = __builtin_amdgcn_wmma_f32_16x16x32_bf16(false, a0, false, b0, (short)0, acc0, false, false);
        acc1 = __builtin_amdgcn_wmma_f32_16x16x32_bf16(false, a1, false, b0, (short)0, acc1, false, false);
      }
      // stage hist_h_projected for pass 2
#pragma unroll
      for (int r = 0; r < 8; ++r)
        xh[(hh * 8 + r) * LROW + ncol] = (bf16)acc1[r];
      __syncthreads();

      float pb = Pb[i * HH + ncol];
      v8f acc2;
#pragma unroll
      for (int r = 0; r < 8; ++r) acc2[r] = pb;
#pragma unroll 4
      for (int k = 0; k < HH; k += 32) {
        v16bf a2 = ldfrag(&xh[nl * LROW + k + hh * 8], &xh[nl * LROW + k + 16 + hh * 8]);
        v16bf b1 = ldfrag(&pwi[k + hh * 8], &pwi[k + 16 + hh * 8]);
        acc2 = __builtin_amdgcn_wmma_f32_16x16x32_bf16(false, a2, false, b1, (short)0, acc2, false, false);
      }

      float hv[8];
#pragma unroll
      for (int r = 0; r < 8; ++r) hv[r] = acc0[r] + acc2[r];
      if (i == 2 || i == 5) {       // residual is x_2(t) in both cases (ref's negative indexing)
        const bf16* res = ring + ((size_t)(2 * 8 + (t & 7)) * BB) * HH;
#pragma unroll
        for (int r = 0; r < 8; ++r)
          hv[r] += (float)res[(size_t)(hh * 8 + r) * HH + ncol];
      }
#pragma unroll
      for (int r = 0; r < 8; ++r) {
        bf16 bv = (bf16)fmaxf(hv[r], 0.0f);
        int m = hh * 8 + r;
        xcur[m * LROW + ncol] = bv;
        if (i + 1 < LL)
          ring[((size_t)((i + 1) * 8 + (t & 7)) * BB + m) * HH + ncol] = bv;
        else
          tops[((size_t)t * BB + m) * HH + ncol] = bv;
      }
      __syncthreads();
    }
  }
}

// ---------------- decoder GEMM: logits = tops @ Dw.T + Db ----------------

__global__ void __launch_bounds__(256) decoder_gemm_kernel(
    const bf16* __restrict__ tops, const bf16* __restrict__ dw,
    const float* __restrict__ Db, float* __restrict__ out) {
  const int lane = threadIdx.x & 31;
  const int wv = threadIdx.x >> 5;        // 0..7
  const int wm = wv & 3, wn = wv >> 2;
  const int nl = lane & 15, hh = lane >> 4;
  const int m0 = (blockIdx.y * 8 + wm * 2) * 16;
  const int n0 = (blockIdx.x * 8 + wn * 4) * 16;

  v8f acc[2][4];
#pragma unroll
  for (int mi = 0; mi < 2; ++mi)
#pragma unroll
    for (int ni = 0; ni < 4; ++ni)
#pragma unroll
      for (int r = 0; r < 8; ++r) acc[mi][ni][r] = 0.0f;

  const bf16* arow0 = tops + (size_t)(m0 + nl) * HH;
  const bf16* arow1 = tops + (size_t)(m0 + 16 + nl) * HH;
  const bf16* brow0 = dw + (size_t)(n0 + 0 * 16 + nl) * HH;
  const bf16* brow1 = dw + (size_t)(n0 + 1 * 16 + nl) * HH;
  const bf16* brow2 = dw + (size_t)(n0 + 2 * 16 + nl) * HH;
  const bf16* brow3 = dw + (size_t)(n0 + 3 * 16 + nl) * HH;
  const bf16* brows[4] = {brow0, brow1, brow2, brow3};

#pragma unroll 2
  for (int k = 0; k < HH; k += 32) {
    v16bf a0 = ldfrag(arow0 + k + hh * 8, arow0 + k + 16 + hh * 8);
    v16bf a1 = ldfrag(arow1 + k + hh * 8, arow1 + k + 16 + hh * 8);
#pragma unroll
    for (int ni = 0; ni < 4; ++ni) {
      v16bf bb = ldfrag(brows[ni] + k + hh * 8, brows[ni] + k + 16 + hh * 8);
      acc[0][ni] = __builtin_amdgcn_wmma_f32_16x16x32_bf16(false, a0, false, bb, (short)0, acc[0][ni], false, false);
      acc[1][ni] = __builtin_amdgcn_wmma_f32_16x16x32_bf16(false, a1, false, bb, (short)0, acc[1][ni], false, false);
    }
  }

#pragma unroll
  for (int ni = 0; ni < 4; ++ni) {
    int n = n0 + ni * 16 + nl;
    if (n < VV) {
      float db = Db[n];
#pragma unroll
      for (int mi = 0; mi < 2; ++mi)
#pragma unroll
        for (int r = 0; r < 8; ++r) {
          int m = m0 + mi * 16 + hh * 8 + r;
          out[(size_t)m * VV + n] = acc[mi][ni][r] + db;
        }
    }
  }
}

// ---------------- per-row log_softmax (in place) ----------------

__global__ void __launch_bounds__(256) logsoftmax_kernel(float* __restrict__ out) {
  __shared__ float red[256];
  float* p = out + (size_t)blockIdx.x * VV;
  float mx = -3.0e38f;
  for (int j = threadIdx.x; j < VV; j += 256) mx = fmaxf(mx, p[j]);
  red[threadIdx.x] = mx;
  __syncthreads();
  for (int s = 128; s > 0; s >>= 1) {
    if (threadIdx.x < s) red[threadIdx.x] = fmaxf(red[threadIdx.x], red[threadIdx.x + s]);
    __syncthreads();
  }
  mx = red[0];
  __syncthreads();
  float sum = 0.0f;
  for (int j = threadIdx.x; j < VV; j += 256) sum += __expf(p[j] - mx);
  red[threadIdx.x] = sum;
  __syncthreads();
  for (int s = 128; s > 0; s >>= 1) {
    if (threadIdx.x < s) red[threadIdx.x] += red[threadIdx.x + s];
    __syncthreads();
  }
  float lse = mx + __logf(red[0]);
  for (int j = threadIdx.x; j < VV; j += 256) p[j] -= lse;
}

// ---------------- history writeback: new_h_i[j] = x_i(T-1-j) ----------------

__global__ void __launch_bounds__(256) hist_out_kernel(
    const bf16* __restrict__ ring, float* __restrict__ outh) {
  int idx = blockIdx.x * 256 + threadIdx.x;
  if (idx >= 21 * BB * HH) return;
  int h = idx & (HH - 1);
  int rem = idx >> 9;
  int b = rem & (BB - 1);
  int slot = rem >> 4;
  int i = 0, j = slot;
  while (j >= i + 1) { j -= (i + 1); i++; }
  int rs = (TT - 1 - j) & 7;
  outh[idx] = (float)ring[(((size_t)i * 8 + rs) * BB + b) * HH + h];
}

// ---------------- launch ----------------

extern "C" void kernel_launch(void* const* d_in, const int* in_sizes, int n_in,
                              void* d_out, int out_size, void* d_ws, size_t ws_size,
                              hipStream_t stream) {
  (void)in_sizes; (void)n_in; (void)out_size; (void)ws_size;
  const int*   tokens = (const int*)  d_in[0];
  const float* embW   = (const float*)d_in[1];
  const float* Cw     = (const float*)d_in[2];
  const float* Cb     = (const float*)d_in[3];
  const float* Pw     = (const float*)d_in[4];
  const float* Pb     = (const float*)d_in[5];
  const float* Dw     = (const float*)d_in[6];
  const float* Db     = (const float*)d_in[7];
  const float* h0 = (const float*)d_in[8];
  const float* h1 = (const float*)d_in[9];
  const float* h2 = (const float*)d_in[10];
  const float* h3 = (const float*)d_in[11];
  const float* h4 = (const float*)d_in[12];
  const float* h5 = (const float*)d_in[13];

  float* out      = (float*)d_out;
  float* out_hist = out + (size_t)TT * BB * VV;

  bf16* dwb   = (bf16*)d_ws;                          // [VPAD][H]
  bf16* cwb   = dwb   + (size_t)VPAD * HH;            // [L][H][H]
  bf16* pwb   = cwb   + (size_t)LL * HH * HH;         // [L][H][H]
  bf16* topsb = pwb   + (size_t)LL * HH * HH;         // [T*B][H]
  bf16* ring  = topsb + (size_t)TT * BB * HH;         // [L][8][B][H]

  {
    int total = 2 * LL * HH * HH;
    prep_w_kernel<<<(total + 255) / 256, 256, 0, stream>>>(Cw, Pw, cwb, pwb);
  }
  {
    size_t total = (size_t)VPAD * HH;
    prep_dw_kernel<<<(unsigned)((total + 255) / 256), 256, 0, stream>>>(Dw, dwb);
  }
  prep_ring_kernel<<<(21 * BB * HH + 255) / 256, 256, 0, stream>>>(h0, h1, h2, h3, h4, h5, ring);

  recurrent_kernel<<<1, 1024, 0, stream>>>(tokens, embW, Cb, Pb, cwb, pwb, ring, topsb);

  decoder_gemm_kernel<<<dim3(VPAD / 128, (TT * BB) / 128), 256, 0, stream>>>(topsb, dwb, Db, out);

  logsoftmax_kernel<<<TT * BB, 256, 0, stream>>>(out);

  hist_out_kernel<<<(21 * BB * HH + 255) / 256, 256, 0, stream>>>(ring, out_hist);
}
